// HungarianMatcher_52527450030676
// MI455X (gfx1250) — compile-verified, matrix operand order checked by reading, not run on hardware
//
#include <hip/hip_runtime.h>
#include <hip/hip_bf16.h>

// ---------------------------------------------------------------------------
// Shapes (fixed by the reference): bs=2, Q=300 -> N=600 queries, n_cls=81,
// H=W=256 -> L=65536 pixels, M=50 targets (padded to 64 for WMMA N-tiles).
// ---------------------------------------------------------------------------
#define NQ        600
#define NQ_PAD    608          // 38 tiles of 16
#define NTILES    38
#define NCLS      81
#define LPIX      65536
#define MT        50
#define MT_PAD    64           // 4 WMMA N-tiles
#define KB_TOTAL  2048         // 65536 / 32 (K per bf16 WMMA)
#define KSPLIT    64           // K-splits per query tile
#define KB_PER    (KB_TOTAL / KSPLIT)

#define L2E 1.4426950408889634f
#define LN2 0.6931471805599453f

// workspace layout (bytes)
#define PM_BYTES   (NQ_PAD * MT_PAD * 4)   // p @ t.T partials
#define FM_BYTES   (NQ_PAD * MT_PAD * 4)   // (pos-neg) @ t.T partials
#define NS_BYTES   (NQ_PAD * 4)            // neg.sum per row
#define PS_BYTES   (NQ_PAD * 4)            // p.sum per row
#define TS_BYTES   (MT_PAD * 4)            // t.sum per target
#define ACC_BYTES  (PM_BYTES + FM_BYTES + NS_BYTES + PS_BYTES + TS_BYTES)
#define TFRAG_OFF  (((ACC_BYTES + 4095) / 4096) * 4096)
#define TFRAG_ELEMS ((size_t)KB_TOTAL * 4 * 32 * 16)   // bf16 elements (8 MB)

typedef __attribute__((ext_vector_type(16))) __bf16          v16bf;
typedef __attribute__((ext_vector_type(16))) unsigned short  u16x16;
typedef __attribute__((ext_vector_type(8)))  float           v8f;

// ---------------------------------------------------------------------------
// Kernel 1: swizzle binary target masks into bf16 WMMA B-fragments.
// Layout: tf[kb][nb][lane][e], e:0..15 maps to K = ks+e (e<8) or 16+ks+(e-8),
// ks = 8*(lane>=16); column n = nb*16 + (lane&15). Matches the ISA 16-bit
// operand lane layout so the main kernel loads one v16bf per lane contiguously.
// ---------------------------------------------------------------------------
__global__ __launch_bounds__(256)
void prep_tfrag_kernel(const int* __restrict__ tgt, unsigned short* __restrict__ tf) {
    unsigned tid = blockIdx.x * 256u + threadIdx.x;
    if (tid >= (unsigned)TFRAG_ELEMS) return;
    int e    =  tid        & 15;
    int lane = (tid >> 4)  & 31;
    int nb   = (tid >> 9)  & 3;
    int kb   =  tid >> 11;
    int n  = nb * 16 + (lane & 15);
    int ks = (lane >> 4) << 3;
    int k  = kb * 32 + ((e < 8) ? (ks + e) : (16 + ks + (e - 8)));
    unsigned short v = 0;
    if (n < MT) v = tgt[(size_t)n * LPIX + k] ? (unsigned short)0x3F80 : (unsigned short)0;
    tf[tid] = v;
}

// ---------------------------------------------------------------------------
// Kernel 2: t.sum per target mask.
// ---------------------------------------------------------------------------
__global__ __launch_bounds__(256)
void tsum_kernel(const int* __restrict__ tgt, float* __restrict__ tsum) {
    __shared__ float sh[256];
    int m = blockIdx.x;
    float s = 0.0f;
    for (int i = threadIdx.x; i < LPIX; i += 256)
        s += (float)tgt[(size_t)m * LPIX + i];
    sh[threadIdx.x] = s;
    __syncthreads();
    for (int st = 128; st > 0; st >>= 1) {
        if ((int)threadIdx.x < st) sh[threadIdx.x] += sh[threadIdx.x + st];
        __syncthreads();
    }
    if (threadIdx.x == 0) tsum[m] = sh[0];
}

// ---------------------------------------------------------------------------
// Kernel 3 (main): one wave per (query tile, K-split).
// Per K-step: load 16x32 f32 logits as an A-fragment, compute sigmoid /
// focal terms, pack p and (pos-neg) to bf16 (native v_cvt, RNE), then
// 8x v_wmma_f32_16x16x32_bf16. Partials combined across splits via
// global f32 atomics.
// ---------------------------------------------------------------------------
__global__ __launch_bounds__(32)
void matcher_main_kernel(const float* __restrict__ x,
                         const unsigned short* __restrict__ tf,
                         float* __restrict__ pm, float* __restrict__ fm,
                         float* __restrict__ ns, float* __restrict__ ps) {
    const int tile  = blockIdx.x;     // 0..37
    const int split = blockIdx.y;     // 0..KSPLIT-1
    const int lane  = threadIdx.x;    // 0..31
    const int mrow  = lane & 15;
    const int half  = (lane >> 4) << 3;   // 0 or 8 (K sub-block)

    int q = tile * 16 + mrow;
    if (q > NQ - 1) q = NQ - 1;           // clamp padded rows (results discarded)
    const float* __restrict__ xrow = x + (size_t)q * LPIX;

    v8f accP[4] = {};   // p @ t.T
    v8f accF[4] = {};   // (pos-neg) @ t.T
    float negs = 0.0f, psum = 0.0f;

    const int kb0 = split * KB_PER;
    for (int kb = kb0; kb < kb0 + KB_PER; ++kb) {
        const int k0 = kb * 32;
        // A-fragment source: K = {half..half+7, 16+half..16+half+7}
        float4 a0 = *(const float4*)(xrow + k0 + half);
        float4 a1 = *(const float4*)(xrow + k0 + half + 4);
        float4 a2 = *(const float4*)(xrow + k0 + 16 + half);
        float4 a3 = *(const float4*)(xrow + k0 + 16 + half + 4);
        float xv[16] = { a0.x, a0.y, a0.z, a0.w,  a1.x, a1.y, a1.z, a1.w,
                         a2.x, a2.y, a2.z, a2.w,  a3.x, a3.y, a3.z, a3.w };

        v16bf pa, fa;
#pragma unroll
        for (int i = 0; i < 16; ++i) {
            float xn  = xv[i];
            float en  = __builtin_amdgcn_exp2f(-xn * L2E);          // e^-x
            float pr  = __builtin_amdgcn_rcpf(1.0f + en);           // sigmoid(x)
            float spn = __builtin_amdgcn_logf(1.0f + en) * LN2;     // softplus(-x)
            float spp = xn + spn;                                   // softplus(x)
            float om  = 1.0f - pr;
            float pos = 0.25f * spn * om * om;                      // ALPHA=0.25, GAMMA=2
            float neg = 0.75f * spp * pr * pr;                      // 1-ALPHA
            negs += neg;
            psum += pr;
            pa[i] = (__bf16)pr;            // native f32->bf16 cvt (RNE)
            fa[i] = (__bf16)(pos - neg);
        }

        const unsigned short* tb = tf + (size_t)kb * (4 * 512) + lane * 16;
#pragma unroll
        for (int nb = 0; nb < 4; ++nb) {
            u16x16 braw = *(const u16x16*)(tb + nb * 512);
            v16bf  b    = __builtin_bit_cast(v16bf, braw);
            accP[nb] = __builtin_amdgcn_wmma_f32_16x16x32_bf16(
                false, pa, false, b, (short)0, accP[nb], false, false);
            accF[nb] = __builtin_amdgcn_wmma_f32_16x16x32_bf16(
                false, fa, false, b, (short)0, accF[nb], false, false);
        }
    }

    // D layout: VGPR r -> row = r + 8*(lane>=16); col = lane&15 (+ nb*16)
    const int rhalf = (lane >> 4) << 3;
#pragma unroll
    for (int nb = 0; nb < 4; ++nb) {
#pragma unroll
        for (int r = 0; r < 8; ++r) {
            int row = tile * 16 + rhalf + r;
            int col = nb * 16 + (lane & 15);
            atomicAdd(&pm[row * MT_PAD + col], accP[nb][r]);
            atomicAdd(&fm[row * MT_PAD + col], accF[nb][r]);
        }
    }
    atomicAdd(&ns[tile * 16 + mrow], negs);
    atomicAdd(&ps[tile * 16 + mrow], psum);
}

// ---------------------------------------------------------------------------
// Kernel 4: compose the final cost matrix C[bs*Q, M].
// ---------------------------------------------------------------------------
__global__ __launch_bounds__(64)
void finalize_kernel(const float* __restrict__ logits, const float* __restrict__ boxes,
                     const float* __restrict__ tboxes, const int* __restrict__ labels,
                     const int* __restrict__ osize,
                     const float* __restrict__ pm, const float* __restrict__ fm,
                     const float* __restrict__ ns, const float* __restrict__ ps,
                     const float* __restrict__ tsum, float* __restrict__ out) {
    const int n = blockIdx.x;          // 0..599
    const int m = threadIdx.x;
    if (m >= MT) return;

    // -softmax(logits)[n, label[m]]
    const float* lg = logits + n * NCLS;
    float mx = lg[0];
    for (int c = 1; c < NCLS; ++c) mx = fmaxf(mx, lg[c]);
    float se = 0.0f;
    for (int c = 0; c < NCLS; ++c) se += __builtin_amdgcn_exp2f((lg[c] - mx) * L2E);
    int lbl = labels[m];
    float cost_class = -__builtin_amdgcn_exp2f((lg[lbl] - mx) * L2E) / se;

    float cost_mask = (fm[n * MT_PAD + m] + ns[n]) * (1.0f / (float)LPIX);
    float cost_dice = 1.0f - (2.0f * pm[n * MT_PAD + m] + 1.0f) / (ps[n] + tsum[m] + 1.0f);

    float Wd = (float)osize[1], Hd = (float)osize[0];
    float tx = tboxes[m * 4 + 0], ty = tboxes[m * 4 + 1];
    float tw = tboxes[m * 4 + 2], th = tboxes[m * 4 + 3];
    float g0 = (tx + tw * 0.5f) / Wd, g1 = (ty + th * 0.5f) / Hd;
    float g2 = tw / Wd, g3 = th / Hd;
    const float* pb = boxes + n * 4;
    float cost_bbox = fabsf(pb[0] - g0) + fabsf(pb[1] - g1)
                    + fabsf(pb[2] - g2) + fabsf(pb[3] - g3);

    out[n * MT + m] = 2.0f * cost_class + 5.0f * cost_mask
                    + 5.0f * cost_dice + 5.0f * cost_bbox;
}

// ---------------------------------------------------------------------------
extern "C" void kernel_launch(void* const* d_in, const int* in_sizes, int n_in,
                              void* d_out, int out_size, void* d_ws, size_t ws_size,
                              hipStream_t stream) {
    const float* pred_logits = (const float*)d_in[0];   // [2,300,81]
    const float* pred_masks  = (const float*)d_in[1];   // [2,300,256,256]
    const float* pred_boxes  = (const float*)d_in[2];   // [2,300,4]
    const float* tgt_boxes   = (const float*)d_in[3];   // [50,4]
    const int*   tgt_labels  = (const int*)d_in[4];     // [50]
    const int*   tgt_masks   = (const int*)d_in[5];     // [50,256,256]
    const int*   orig_size   = (const int*)d_in[6];     // [2]
    (void)in_sizes; (void)n_in; (void)out_size; (void)ws_size;

    char* ws = (char*)d_ws;
    float* pm   = (float*)(ws);
    float* fm   = (float*)(ws + PM_BYTES);
    float* nsum = (float*)(ws + PM_BYTES + FM_BYTES);
    float* psum = (float*)(ws + PM_BYTES + FM_BYTES + NS_BYTES);
    float* tsum = (float*)(ws + PM_BYTES + FM_BYTES + NS_BYTES + PS_BYTES);
    unsigned short* tfrag = (unsigned short*)(ws + TFRAG_OFF);

    // zero atomic-accumulation region each call (graph-capture safe)
    hipMemsetAsync(ws, 0, ACC_BYTES, stream);

    prep_tfrag_kernel<<<(unsigned)((TFRAG_ELEMS + 255) / 256), 256, 0, stream>>>(
        tgt_masks, tfrag);
    tsum_kernel<<<MT, 256, 0, stream>>>(tgt_masks, tsum);

    dim3 grid(NTILES, KSPLIT);
    matcher_main_kernel<<<grid, 32, 0, stream>>>(pred_masks, tfrag, pm, fm, nsum, psum);

    finalize_kernel<<<NQ, 64, 0, stream>>>(pred_logits, pred_boxes, tgt_boxes,
                                           tgt_labels, orig_size,
                                           pm, fm, nsum, psum, tsum, (float*)d_out);
}
